// BaseConv2d_55456617726338
// MI455X (gfx1250) — matrix operand order repaired and checked by reading, not compiled, not run on hardware
//
#include <hip/hip_runtime.h>
#include <hip/hip_bf16.h>

// ---------------------------------------------------------------------------
// BaseConv2d (3x3, s1, p1) with per-(b,c) input conditioning, weight scale,
// and reference-exact bias indexing, as implicit GEMM on CDNA5 WMMA (bf16),
// staged through global_load_async_to_lds_b128 (ASYNCcnt pipeline).
//
// Single workspace base (one SGPR pair for all async loads):
//   [XOFF=0,          134217728)  input bf16 NHWC, cond folded in
//   [WOFF=134217728, +1179648  )  weights bf16 [tap][oc][ic], scaled
//   [ZOFF=135397376, +256      )  zero pad region (conv halo source)
// ---------------------------------------------------------------------------

typedef __attribute__((ext_vector_type(16))) __bf16 v16bf;
typedef __attribute__((ext_vector_type(8)))  float  v8f;

#define IN_CH  256
#define OUT_CH 256
#define BATCH  16
#define HH     128
#define WW     128

#define BM 128
#define BN 64
#define KITERS 72            // 9 taps * (256/32)
#define LDA 40               // LDS row stride in halves (80B, conflict-free)

#define WOFF 134217728       // byte offset of weights in ws
#define ZOFF 135397376       // byte offset of zero pad in ws

static __device__ __forceinline__ unsigned short f2bf(float f) {
  unsigned int u = __float_as_uint(f);
  u += 0x7FFFu + ((u >> 16) & 1u);          // round-to-nearest-even
  return (unsigned short)(u >> 16);
}

// gfx1250 async copy: 16B per lane, global (saddr base + vgpr byte offset) -> LDS
static __device__ __forceinline__ void async_ld16(unsigned ldsOff, int gOff,
                                                  const void* base) {
  asm volatile("global_load_async_to_lds_b128 %0, %1, %2 offset:0"
               :: "v"(ldsOff), "v"(gOff), "s"(base) : "memory");
}

// --- pre-pass 1: NCHW f32 * cond[b,c] -> NHWC bf16 (LDS-tiled transpose) ---
__global__ void cond_nchw_to_nhwc_bf16(const float* __restrict__ in,
                                       const float* __restrict__ cond,
                                       unsigned short* __restrict__ x) {
  __shared__ float tile[32][33];
  const int t  = threadIdx.x;
  const int bx = blockIdx.x;
  const int wb = bx & 3;            // W/32
  const int cb = (bx >> 2) & 7;     // C/32
  const int h  = (bx >> 5) & 127;   // H
  const int bb = bx >> 12;          // B
  const int w0 = wb * 32, c0 = cb * 32;

  {  // coalesced read along W
    const int wi = t & 31, cr = t >> 5;
#pragma unroll
    for (int k = 0; k < 4; ++k) {
      const int ci = cr + k * 8;
      tile[ci][wi] = in[((size_t)(bb * IN_CH + c0 + ci) * HH + h) * WW + w0 + wi];
    }
  }
  __syncthreads();
  {  // coalesced write along C, cond folded in
    const int ci = t & 31, wr = t >> 5;
    const float cf = cond[bb * IN_CH + c0 + ci];
#pragma unroll
    for (int k = 0; k < 4; ++k) {
      const int wi = wr + k * 8;
      x[((size_t)(bb * HH + h) * WW + w0 + wi) * IN_CH + c0 + ci] =
          f2bf(tile[ci][wi] * cf);
    }
  }
}

// --- pre-pass 2: weight OIHW f32 * (1/48) -> bf16 [tap][oc][ic]; also zero
// --- the 256B halo pad region (rewritten every launch: deterministic). -----
__global__ void weight_to_bf16(const float* __restrict__ w,
                               unsigned short* __restrict__ wv,
                               uint4* __restrict__ zeropad) {
  const int idx = blockIdx.x * blockDim.x + threadIdx.x;  // ic fastest
  if (blockIdx.x == 0 && threadIdx.x < 16)
    zeropad[threadIdx.x] = make_uint4(0u, 0u, 0u, 0u);
  if (idx >= 9 * OUT_CH * IN_CH) return;
  const int ic  = idx & 255;
  const int oc  = (idx >> 8) & 255;
  const int tap = idx >> 16;
  const float scale = 0.02083333333333333f;               // 1/sqrt(256*9)
  wv[((size_t)tap * OUT_CH + oc) * IN_CH + ic] =
      f2bf(w[(size_t)(oc * IN_CH + ic) * 9 + tap] * scale);
}

// --- main: implicit-GEMM conv, async-to-LDS staging + v_wmma bf16 ----------
__global__ __launch_bounds__(256)
void conv_wmma(const char* __restrict__ ws,            // single async base
               const float* __restrict__ bias,
               float* __restrict__ out) {              // [B][OC][H][W] f32
  __shared__ __align__(16) unsigned short As[2][BM][LDA];
  __shared__ __align__(16) unsigned short Bs[2][BN][LDA];

  const int t    = threadIdx.x;
  const int lane = t & 31;
  const int wave = t >> 5;
  const int wm   = wave & 3;        // 4 waves along M
  const int wn   = wave >> 2;       // 2 waves along N
  const int mOff = wm * 32;
  const int nOff = wn * 32;

  const int s      = blockIdx.x;          // 16*128*2 = 4096 spatial tiles
  const int owBase = (s & 1) * BN;
  const int oh     = (s >> 1) & 127;
  const int bb     = s >> 8;
  const int ocBase = blockIdx.y * BM;

  // staging roles: B tile 64 rows x 4 chunks; A tile 128 rows x 4 (2/thread)
  const int nRow  = t >> 2;
  const int chunk = t & 3;
  const int aRow0 = t >> 2;
  const int aRow1 = aRow0 + 64;

  // LDS destination byte offsets (workgroup-relative low 32 bits)
  unsigned ldsB[2], ldsA0[2], ldsA1[2];
#pragma unroll
  for (int p = 0; p < 2; ++p) {
    ldsB[p]  = (unsigned)(uintptr_t)&Bs[p][nRow][chunk * 8];
    ldsA0[p] = (unsigned)(uintptr_t)&As[p][aRow0][chunk * 8];
    ldsA1[p] = (unsigned)(uintptr_t)&As[p][aRow1][chunk * 8];
  }

  auto issue = [&](int it, int nb) {
    const int tap = it >> 3;
    const int kc  = (it & 7) << 5;
    const int kh  = tap / 3;
    const int kw  = tap - kh * 3;
    const int ih  = oh + kh - 1;
    const int iw  = owBase + nRow + kw - 1;
    const bool vld = ((unsigned)ih < (unsigned)HH) && ((unsigned)iw < (unsigned)WW);
    int boff = ((((bb * HH + ih) * WW + iw) * IN_CH + kc) << 1) + chunk * 16;
    boff = vld ? boff : (int)ZOFF;                        // halo -> zeros
    const int aoff0 = WOFF + ((((tap * OUT_CH + ocBase + aRow0) * IN_CH) + kc) << 1) + chunk * 16;
    const int aoff1 = WOFF + ((((tap * OUT_CH + ocBase + aRow1) * IN_CH) + kc) << 1) + chunk * 16;
    async_ld16(ldsB[nb],  boff,  ws);
    async_ld16(ldsA0[nb], aoff0, ws);
    async_ld16(ldsA1[nb], aoff1, ws);
  };

  issue(0, 0);                       // prologue: tile 0 -> buffer 0

  v8f acc[2][2] = {};
  const int mrow = lane & 15;
  const int hi   = lane >> 4;
  union Frag { uint4 u[2]; v16bf v; };

  int buf = 0;
  for (int it = 0; it < KITERS; ++it) {
    const int nb = buf ^ 1;
    if (it + 1 < KITERS) {
      issue(it + 1, nb);             // prefetch next tile (3 async ops)
      asm volatile("s_wait_asynccnt 0x3" ::: "memory");   // front tile done
    } else {
      asm volatile("s_wait_asynccnt 0x0" ::: "memory");
    }
    __syncthreads();                 // all waves' front tile visible

    // A frags (ISA 16-bit A 16x32): lane<16 K[0..7]+K[16..23]; hi K[8..15]+K[24..31]
    Frag a[2], b[2];
#pragma unroll
    for (int i = 0; i < 2; ++i) {
      const unsigned short* p = &As[buf][mOff + i * 16 + mrow][hi * 8];
      a[i].u[0] = *(const uint4*)p;
      a[i].u[1] = *(const uint4*)(p + 16);
    }
    // B frags: lane<16 col n K[0..7]+K[16..23]; hi K[8..15]+K[24..31]
#pragma unroll
    for (int j = 0; j < 2; ++j) {
      const unsigned short* p = &Bs[buf][nOff + j * 16 + mrow][hi * 8];
      b[j].u[0] = *(const uint4*)p;
      b[j].u[1] = *(const uint4*)(p + 16);
    }
#pragma unroll
    for (int i = 0; i < 2; ++i)
#pragma unroll
      for (int j = 0; j < 2; ++j)
        acc[i][j] = __builtin_amdgcn_wmma_f32_16x16x32_bf16(
            false, a[i].v, false, b[j].v, (short)0, acc[i][j], false, false);

    __syncthreads();                 // reads done before buf is re-filled
    buf = nb;
  }

  // epilogue: C layout -> lane 0-15: M=r,N=lane ; lane 16-31: M=8+r,N=lane-16
#pragma unroll
  for (int i = 0; i < 2; ++i)
#pragma unroll
    for (int j = 0; j < 2; ++j)
#pragma unroll
      for (int r = 0; r < 8; ++r) {
        const int m  = mOff + i * 16 + r + 8 * hi;
        const int n  = nOff + j * 16 + (lane & 15);
        const int oc = ocBase + m;
        const int ow = owBase + n;
        // reference: bias_mat[i,j] = bias[(i*OUT_CH + j) // B]
        const float bsum = bias[(bb * OUT_CH + oc) / BATCH];
        out[((size_t)(bb * OUT_CH + oc) * HH + oh) * WW + ow] = acc[i][j][r] + bsum;
      }
}

extern "C" void kernel_launch(void* const* d_in, const int* in_sizes, int n_in,
                              void* d_out, int out_size, void* d_ws, size_t ws_size,
                              hipStream_t stream) {
  const float* input  = (const float*)d_in[0];
  const float* cond   = (const float*)d_in[1];
  const float* weight = (const float*)d_in[2];
  const float* bias   = (const float*)d_in[3];
  float* out = (float*)d_out;

  char* ws = (char*)d_ws;
  unsigned short* wsX = (unsigned short*)ws;            // XOFF = 0
  unsigned short* wsW = (unsigned short*)(ws + WOFF);
  uint4*          wsZ = (uint4*)(ws + ZOFF);

  hipLaunchKernelGGL(cond_nchw_to_nhwc_bf16, dim3(65536), dim3(256), 0, stream,
                     input, cond, wsX);
  hipLaunchKernelGGL(weight_to_bf16, dim3((9 * OUT_CH * IN_CH) / 256), dim3(256),
                     0, stream, weight, wsW, wsZ);
  hipLaunchKernelGGL(conv_wmma, dim3(4096, 2), dim3(256), 0, stream,
                     ws, bias, out);
}
